// CrossAttn__53592601919505
// MI455X (gfx1250) — matrix operand order repaired and checked by reading, not compiled, not run on hardware
//
#include <hip/hip_runtime.h>
#include <hip/hip_bf16.h>

typedef __bf16 bf16_t;
typedef __attribute__((ext_vector_type(16))) __bf16 v16bf;
typedef __attribute__((ext_vector_type(8)))  float  v8f;
typedef __attribute__((ext_vector_type(4)))  int    i32x4_t;
typedef __attribute__((address_space(1))) i32x4_t* as1_i32x4_p;
typedef __attribute__((address_space(3))) i32x4_t* as3_i32x4_p;

#define B_     4
#define NTOK   2048
#define CH     512
#define ROWS   (B_ * NTOK)   /* 8192 */
#define HEADS_ 8
#define SCALE_ 0.125f        /* 64^-0.5 */

#if __has_builtin(__builtin_amdgcn_global_load_async_to_lds_b128)
#define HAVE_ASYNC_LDS 1
__device__ __forceinline__ void async_ld_b128(const bf16_t* g, bf16_t* l) {
  __builtin_amdgcn_global_load_async_to_lds_b128(
      (as1_i32x4_p)g, (as3_i32x4_p)l, 0, 0);
}
#else
#define HAVE_ASYNC_LDS 0
#endif

__device__ __forceinline__ void wait_async0() {
#if __has_builtin(__builtin_amdgcn_s_wait_asynccnt)
  __builtin_amdgcn_s_wait_asynccnt(0);
#else
  asm volatile("s_wait_asynccnt 0x0" ::: "memory");
#endif
}

__device__ __forceinline__ v8f v8f_zero() {
  v8f z;
#pragma unroll
  for (int i = 0; i < 8; ++i) z[i] = 0.f;
  return z;
}

// A-fragment 16x32 bf16: m = lane&15, half = lane>>4.
// element e -> K = (e/8)*16 + half*8 + (e%8). Two contiguous 16B chunks.
__device__ __forceinline__ v16bf load_frag_a(const bf16_t* base, int ld,
                                             int row0, int k0, int lane) {
  int m = lane & 15, half = lane >> 4;
  const bf16_t* p = base + (size_t)(row0 + m) * ld + k0 + half * 8;
  union { uint4 u[2]; v16bf v; } f;
  f.u[0] = *(const uint4*)(p);
  f.u[1] = *(const uint4*)(p + 16);
  return f.v;
}

// B-fragment 32x16 bf16 from column-major-stored B (Bt[N][K]):
// n = lane&15, element e -> K = (e/8)*16 + (lane>>4)*8 + (e%8). Two 16B chunks.
__device__ __forceinline__ v16bf load_frag_b(const bf16_t* baseT, int ld,
                                             int col0, int k0, int lane) {
  int n = lane & 15, kh = lane >> 4;
  const bf16_t* p = baseT + (size_t)(col0 + n) * ld + k0 + kh * 8;
  union { uint4 u[2]; v16bf v; } f;
  f.u[0] = *(const uint4*)(p);
  f.u[1] = *(const uint4*)(p + 16);
  return f.v;
}

__device__ __forceinline__ v8f wmma_bf16(v16bf a, v16bf b, v8f c) {
  return __builtin_amdgcn_wmma_f32_16x16x32_bf16(false, a, false, b,
                                                 (short)0, c, false, false);
}

__device__ __forceinline__ float gelu_f(float x) {
  float t = tanhf(0.7978845608028654f * (x + 0.044715f * x * x * x));
  return 0.5f * x * (1.0f + t);
}

// ---------------- weight transpose + fp32->bf16 ----------------
// W [K,N] row-major fp32  ->  Wt [N,K] row-major bf16
__global__ __launch_bounds__(256) void wt_kernel(const float* __restrict__ W,
                                                 bf16_t* __restrict__ Wt,
                                                 int K, int N) {
  long idx = (long)blockIdx.x * 256 + threadIdx.x;
  if (idx >= (long)K * N) return;
  int n = (int)(idx / K), k = (int)(idx % K);
  Wt[idx] = (bf16_t)W[(size_t)k * N + n];
}

// ---------------- LayerNorm: one wave per 512-wide row -> bf16 -------------
__global__ __launch_bounds__(256) void ln_rows_kernel(
    const float* __restrict__ src, const float* __restrict__ gamma,
    const float* __restrict__ beta, bf16_t* __restrict__ dst, int nrows) {
  int lane = threadIdx.x & 31;
  int row = blockIdx.x * 8 + (threadIdx.x >> 5);
  if (row >= nrows) return;
  const float4* p = (const float4*)(src + (size_t)row * CH);
  float4 v[4];
  float s = 0.f, ss = 0.f;
#pragma unroll
  for (int c = 0; c < 4; ++c) {
    v[c] = p[lane + c * 32];
    s  += v[c].x + v[c].y + v[c].z + v[c].w;
    ss += v[c].x * v[c].x + v[c].y * v[c].y + v[c].z * v[c].z + v[c].w * v[c].w;
  }
#pragma unroll
  for (int off = 16; off >= 1; off >>= 1) {
    s  += __shfl_xor(s, off, 32);
    ss += __shfl_xor(ss, off, 32);
  }
  float mean = s * (1.f / CH);
  float var  = ss * (1.f / CH) - mean * mean;
  float rstd = rsqrtf(var + 1e-5f);
  bf16_t* d = dst + (size_t)row * CH;
#pragma unroll
  for (int c = 0; c < 4; ++c) {
    int i4 = lane + c * 32;
    float4 g = ((const float4*)gamma)[i4];
    float4 b = ((const float4*)beta)[i4];
    int idx = i4 * 4;
    d[idx + 0] = (bf16_t)((v[c].x - mean) * rstd * g.x + b.x);
    d[idx + 1] = (bf16_t)((v[c].y - mean) * rstd * g.y + b.y);
    d[idx + 2] = (bf16_t)((v[c].z - mean) * rstd * g.z + b.z);
    d[idx + 3] = (bf16_t)((v[c].w - mean) * rstd * g.w + b.w);
  }
}

// ---------------- LDS-staged bf16 GEMM with fused epilogues ----------------
enum { E_QPROJ = 0, E_KVPROJ = 1, E_LINRES = 2, E_FFN1 = 3 };

template <int MODE>
__global__ __launch_bounds__(256) void gemm_kernel(
    const bf16_t* __restrict__ A, const bf16_t* __restrict__ Bt,
    const float* __restrict__ bias, const float* __restrict__ res,
    float* __restrict__ outf, bf16_t* __restrict__ outb0,
    bf16_t* __restrict__ outb1, int N, int K) {
  int tid = threadIdx.x, lane = tid & 31, wave = tid >> 5;
  int wm = wave & 3, wn = wave >> 2;         // 4x2 wave grid -> 32x64 per wave
  int bm = blockIdx.y * 128, bn = blockIdx.x * 128;
  int lr = tid >> 1, lc = (tid & 1) * 16;    // 2 threads per 32-elem row
  const bf16_t* ga = A  + (size_t)(bm + lr) * K + lc;
  const bf16_t* gb = Bt + (size_t)(bn + lr) * K + lc;

  v8f acc[2][4];
#pragma unroll
  for (int mt = 0; mt < 2; ++mt)
#pragma unroll
    for (int nt = 0; nt < 4; ++nt) acc[mt][nt] = v8f_zero();

#if HAVE_ASYNC_LDS
  // Double-buffered: global -> LDS via ASYNCcnt-tracked direct path.
  __shared__ __align__(16) bf16_t sA[2][128][32];
  __shared__ __align__(16) bf16_t sB[2][128][32];
  auto prefetch = [&](int k0, int buf) {
    async_ld_b128(ga + k0,     &sA[buf][lr][lc]);
    async_ld_b128(ga + k0 + 8, &sA[buf][lr][lc + 8]);
    async_ld_b128(gb + k0,     &sB[buf][lr][lc]);
    async_ld_b128(gb + k0 + 8, &sB[buf][lr][lc + 8]);
  };
  prefetch(0, 0);
  wait_async0();
  __syncthreads();
  int nsteps = K >> 5;
  for (int i = 0; i < nsteps; ++i) {
    int cur = i & 1;
    if (i + 1 < nsteps) prefetch((i + 1) * 32, cur ^ 1);
    v16bf af[2], bfr[4];
#pragma unroll
    for (int mt = 0; mt < 2; ++mt)
      af[mt] = load_frag_a(&sA[cur][0][0], 32, wm * 32 + mt * 16, 0, lane);
#pragma unroll
    for (int nt = 0; nt < 4; ++nt)
      bfr[nt] = load_frag_b(&sB[cur][0][0], 32, wn * 64 + nt * 16, 0, lane);
#pragma unroll
    for (int mt = 0; mt < 2; ++mt)
#pragma unroll
      for (int nt = 0; nt < 4; ++nt)
        acc[mt][nt] = wmma_bf16(af[mt], bfr[nt], acc[mt][nt]);
    if (i + 1 < nsteps) {
      wait_async0();
      __syncthreads();
    }
  }
#else
  // Fallback: register-staged single-buffer pipeline.
  __shared__ __align__(16) bf16_t sA[128][32];
  __shared__ __align__(16) bf16_t sB[128][32];
  for (int k0 = 0; k0 < K; k0 += 32) {
    uint4 a0 = *(const uint4*)(ga + k0);
    uint4 a1 = *(const uint4*)(ga + k0 + 8);
    uint4 b0 = *(const uint4*)(gb + k0);
    uint4 b1 = *(const uint4*)(gb + k0 + 8);
    __syncthreads();
    *(uint4*)&sA[lr][lc]     = a0;
    *(uint4*)&sA[lr][lc + 8] = a1;
    *(uint4*)&sB[lr][lc]     = b0;
    *(uint4*)&sB[lr][lc + 8] = b1;
    __syncthreads();
    v16bf af[2], bfr[4];
#pragma unroll
    for (int mt = 0; mt < 2; ++mt)
      af[mt] = load_frag_a(&sA[0][0], 32, wm * 32 + mt * 16, 0, lane);
#pragma unroll
    for (int nt = 0; nt < 4; ++nt)
      bfr[nt] = load_frag_b(&sB[0][0], 32, wn * 64 + nt * 16, 0, lane);
#pragma unroll
    for (int mt = 0; mt < 2; ++mt)
#pragma unroll
      for (int nt = 0; nt < 4; ++nt)
        acc[mt][nt] = wmma_bf16(af[mt], bfr[nt], acc[mt][nt]);
  }
#endif

  int nn = lane & 15, hf = lane >> 4;
#pragma unroll
  for (int mt = 0; mt < 2; ++mt) {
#pragma unroll
    for (int nt = 0; nt < 4; ++nt) {
      int wrow0 = bm + wm * 32 + mt * 16;
      int wcol0 = bn + wn * 64 + nt * 16;
      int gcol = wcol0 + nn;
#pragma unroll
      for (int r = 0; r < 8; ++r) {
        int grow = wrow0 + r + 8 * hf;       // C layout: M = r + 8*(lane/16)
        float vacc = acc[mt][nt][r];
        if constexpr (MODE == E_QPROJ) {
          int b = grow >> 11, tok = grow & 2047;
          int h = gcol >> 6, d = gcol & 63;
          outb0[(((size_t)(b * 8 + h)) * NTOK + tok) * 64 + d] = (bf16_t)vacc;
        } else if constexpr (MODE == E_KVPROJ) {
          int b = grow >> 11, tok = grow & 2047;
          if (gcol < 512) {                  // K: [b,h,tok,d]
            int h = gcol >> 6, d = gcol & 63;
            outb0[(((size_t)(b * 8 + h)) * NTOK + tok) * 64 + d] = (bf16_t)vacc;
          } else {                           // V transposed: [b,h,d,tok]
            int c2 = gcol - 512;
            int h = c2 >> 6, d = c2 & 63;
            outb1[(((size_t)(b * 8 + h)) * 64 + d) * NTOK + tok] = (bf16_t)vacc;
          }
        } else if constexpr (MODE == E_LINRES) {
          size_t o = (size_t)grow * N + gcol;
          outf[o] = vacc + bias[gcol] + res[o];
        } else {                             // E_FFN1: gelu -> bf16
          outb0[(size_t)grow * N + gcol] = (bf16_t)gelu_f(vacc + bias[gcol]);
        }
      }
    }
  }
}

// ---------------- flash attention, 1 wave = 16 q rows, Bc = 64 -------------
__global__ __launch_bounds__(256) void attn_kernel(
    const bf16_t* __restrict__ Q, const bf16_t* __restrict__ Kb,
    const bf16_t* __restrict__ Vt, bf16_t* __restrict__ O) {
  __shared__ __align__(16) bf16_t sP[8][16 * 64];  // per-wave P bounce buffer
  int lane = threadIdx.x & 31, wave = threadIdx.x >> 5;
  int bh = blockIdx.y;                             // b*8 + h
  int qrow0 = blockIdx.x * 128 + wave * 16;
  const bf16_t* Qp = Q  + (size_t)bh * NTOK * 64;
  const bf16_t* Kp = Kb + (size_t)bh * NTOK * 64;
  const bf16_t* Vp = Vt + (size_t)bh * 64 * NTOK;  // [d][tok]

  v16bf qa0 = load_frag_a(Qp, 64, qrow0, 0, lane);
  v16bf qa1 = load_frag_a(Qp, 64, qrow0, 32, lane);
  v8f o[4];
#pragma unroll
  for (int t = 0; t < 4; ++t) o[t] = v8f_zero();
  float m_r[8], l_r[8];
#pragma unroll
  for (int r = 0; r < 8; ++r) { m_r[r] = -1e30f; l_r[r] = 0.f; }

  bf16_t* ps = &sP[wave][0];
  int nn = lane & 15, hf = lane >> 4;

  for (int j0 = 0; j0 < NTOK; j0 += 64) {
    // S = Q (16x64) x K^T (64x64): K stored [tok][d] => contiguous B frags
    v8f s[4];
#pragma unroll
    for (int t = 0; t < 4; ++t) {
      v16bf kb0 = load_frag_b(Kp, 64, j0 + t * 16, 0,  lane);
      v16bf kb1 = load_frag_b(Kp, 64, j0 + t * 16, 32, lane);
      s[t] = wmma_bf16(qa0, kb0, v8f_zero());
      s[t] = wmma_bf16(qa1, kb1, s[t]);
    }

    // online softmax: row = r + 8*hf lives in 16 lanes of one half-wave
#pragma unroll
    for (int r = 0; r < 8; ++r) {
      float a0 = s[0][r] * SCALE_;
      float a1 = s[1][r] * SCALE_;
      float a2 = s[2][r] * SCALE_;
      float a3 = s[3][r] * SCALE_;
      float mx = fmaxf(fmaxf(a0, a1), fmaxf(a2, a3));
#pragma unroll
      for (int off = 1; off < 16; off <<= 1)
        mx = fmaxf(mx, __shfl_xor(mx, off, 32));
      float mnew = fmaxf(m_r[r], mx);
      float scl = __expf(m_r[r] - mnew);
      float p0 = __expf(a0 - mnew);
      float p1 = __expf(a1 - mnew);
      float p2 = __expf(a2 - mnew);
      float p3 = __expf(a3 - mnew);
      float rs = (p0 + p1) + (p2 + p3);
#pragma unroll
      for (int off = 1; off < 16; off <<= 1) rs += __shfl_xor(rs, off, 32);
      l_r[r] = l_r[r] * scl + rs;
      m_r[r] = mnew;
#pragma unroll
      for (int t = 0; t < 4; ++t) o[t][r] *= scl;
      int rowb = (r + 8 * hf) * 64 + nn;
      ps[rowb]      = (bf16_t)p0;            // C-layout -> LDS
      ps[rowb + 16] = (bf16_t)p1;
      ps[rowb + 32] = (bf16_t)p2;
      ps[rowb + 48] = (bf16_t)p3;
    }
    asm volatile("" ::: "memory");  // keep DS store->load order (HW in-order)
    v16bf pa0 = load_frag_a(ps, 64, 0, 0,  lane);    // reread in A-layout
    v16bf pa1 = load_frag_a(ps, 64, 0, 32, lane);

    // O += P (16x64) x V (64x64): Vt[d][tok] => contiguous B frags
#pragma unroll
    for (int t = 0; t < 4; ++t) {
      v16bf vb0 = load_frag_b(Vp, NTOK, t * 16, j0,      lane);
      v16bf vb1 = load_frag_b(Vp, NTOK, t * 16, j0 + 32, lane);
      o[t] = wmma_bf16(pa0, vb0, o[t]);
      o[t] = wmma_bf16(pa1, vb1, o[t]);
    }
  }

  int b = bh >> 3, h = bh & 7;
#pragma unroll
  for (int r = 0; r < 8; ++r) {
    int i = qrow0 + r + 8 * hf;
    float inv = 1.f / l_r[r];
    size_t base = ((size_t)b * NTOK + i) * 512 + h * 64;
#pragma unroll
    for (int t = 0; t < 4; ++t)
      O[base + t * 16 + nn] = (bf16_t)(o[t][r] * inv);
  }
}

// ---------------------------------------------------------------------------
extern "C" void kernel_launch(void* const* d_in, const int* in_sizes, int n_in,
                              void* d_out, int out_size, void* d_ws,
                              size_t ws_size, hipStream_t stream) {
  const float* x   = (const float*)d_in[0];
  const float* y   = (const float*)d_in[1];
  const float* Wq  = (const float*)d_in[2];
  const float* Wkv = (const float*)d_in[3];
  const float* Wp  = (const float*)d_in[4];
  const float* bp  = (const float*)d_in[5];
  const float* W1  = (const float*)d_in[6];
  const float* b1  = (const float*)d_in[7];
  const float* W2  = (const float*)d_in[8];
  const float* b2  = (const float*)d_in[9];
  const float* g1y = (const float*)d_in[10];
  const float* b1y = (const float*)d_in[11];
  const float* g1x = (const float*)d_in[12];
  const float* b1x = (const float*)d_in[13];
  const float* g2  = (const float*)d_in[14];
  const float* b2n = (const float*)d_in[15];
  float* out = (float*)d_out;

  char* w = (char*)d_ws;
  auto alloc = [&](size_t bytes) {
    char* p = w;
    w += (bytes + 255) & ~(size_t)255;
    return p;
  };
  bf16_t* xln  = (bf16_t*)alloc((size_t)ROWS * CH * 2);
  bf16_t* yln  = (bf16_t*)alloc((size_t)ROWS * CH * 2);
  bf16_t* WqT  = (bf16_t*)alloc((size_t)512 * 512 * 2);
  bf16_t* WkvT = (bf16_t*)alloc((size_t)1024 * 512 * 2);
  bf16_t* WpT  = (bf16_t*)alloc((size_t)512 * 512 * 2);
  bf16_t* W1T  = (bf16_t*)alloc((size_t)2048 * 512 * 2);
  bf16_t* W2T  = (bf16_t*)alloc((size_t)512 * 2048 * 2);
  bf16_t* qb   = (bf16_t*)alloc((size_t)ROWS * 512 * 2);
  bf16_t* kb   = (bf16_t*)alloc((size_t)ROWS * 512 * 2);
  bf16_t* vtb  = (bf16_t*)alloc((size_t)ROWS * 512 * 2);
  bf16_t* ob   = (bf16_t*)alloc((size_t)ROWS * 512 * 2);
  float*  ores = (float*) alloc((size_t)ROWS * 512 * 4);
  bf16_t* hb   = (bf16_t*)alloc((size_t)ROWS * 512 * 2);
  bf16_t* m1   = (bf16_t*)alloc((size_t)ROWS * 2048 * 2);

  // weights -> transposed bf16
  wt_kernel<<<(512 * 512 + 255) / 256, 256, 0, stream>>>(Wq, WqT, 512, 512);
  wt_kernel<<<(512 * 1024 + 255) / 256, 256, 0, stream>>>(Wkv, WkvT, 512, 1024);
  wt_kernel<<<(512 * 512 + 255) / 256, 256, 0, stream>>>(Wp, WpT, 512, 512);
  wt_kernel<<<(512 * 2048 + 255) / 256, 256, 0, stream>>>(W1, W1T, 512, 2048);
  wt_kernel<<<(2048 * 512 + 255) / 256, 256, 0, stream>>>(W2, W2T, 2048, 512);

  // LN1
  ln_rows_kernel<<<ROWS / 8, 256, 0, stream>>>(x, g1x, b1x, xln, ROWS);
  ln_rows_kernel<<<ROWS / 8, 256, 0, stream>>>(y, g1y, b1y, yln, ROWS);

  // projections
  gemm_kernel<E_QPROJ><<<dim3(512 / 128, ROWS / 128), 256, 0, stream>>>(
      yln, WqT, nullptr, nullptr, nullptr, qb, nullptr, 512, 512);
  gemm_kernel<E_KVPROJ><<<dim3(1024 / 128, ROWS / 128), 256, 0, stream>>>(
      xln, WkvT, nullptr, nullptr, nullptr, kb, vtb, 1024, 512);

  // attention
  attn_kernel<<<dim3(NTOK / 128, B_ * HEADS_), 256, 0, stream>>>(qb, kb, vtb, ob);

  // out proj + residual(y)
  gemm_kernel<E_LINRES><<<dim3(512 / 128, ROWS / 128), 256, 0, stream>>>(
      ob, WpT, bp, y, ores, nullptr, nullptr, 512, 512);

  // LN2 + MLP + residual(o)
  ln_rows_kernel<<<ROWS / 8, 256, 0, stream>>>(ores, g2, b2n, hb, ROWS);
  gemm_kernel<E_FFN1><<<dim3(2048 / 128, ROWS / 128), 256, 0, stream>>>(
      hb, W1T, b1, nullptr, nullptr, m1, nullptr, 2048, 512);
  gemm_kernel<E_LINRES><<<dim3(512 / 128, ROWS / 128), 256, 0, stream>>>(
      m1, W2T, b2, ores, out, nullptr, nullptr, 512, 2048);
}
